// QwenDoubleStreamAttention_30494267802037
// MI455X (gfx1250) — matrix operand order repaired and checked by reading, not compiled, not run on hardware
//
#include <hip/hip_runtime.h>

// ---------------------------------------------------------------------------
// Qwen double-stream attention for MI455X (gfx1250, wave32, WMMA).
// Matmuls on v_wmma_f32_16x16x32_f16 (f16 in, f32 accumulate); straight-copy
// LDS tiles staged with global_load_async_to_lds_b128 (ASYNCcnt path).
// ---------------------------------------------------------------------------

typedef __attribute__((ext_vector_type(16))) _Float16 v16h;
typedef __attribute__((ext_vector_type(2)))  _Float16 h2;
typedef __attribute__((ext_vector_type(8)))  float    v8f;

#define DIM    3072
#define HD     128
#define NH     24
#define S_IMG  2048
#define S_TXT  256
#define S_ALL  2304   // txt rows 0..255, img rows 256..2303 (reference concat order)

// Load a v16h fragment as two 16-byte chunks (LDS or global, 16B aligned).
static __device__ __forceinline__ v16h ld_frag(const _Float16* lo, const _Float16* hi) {
    v16h r;
    ((int4*)&r)[0] = *(const int4*)lo;
    ((int4*)&r)[1] = *(const int4*)hi;
    return r;
}

// Async DMA: 16 bytes global -> LDS, no VGPR staging (tracked by ASYNCcnt).
// Flat generic pointers to __shared__ have the LDS byte offset in addr[31:0].
static __device__ __forceinline__ void async_b128(unsigned lds_addr, unsigned long long gaddr) {
    asm volatile("global_load_async_to_lds_b128 %0, %1, off"
                 :: "v"(lds_addr), "v"(gaddr) : "memory");
}
static __device__ __forceinline__ void wait_async0() {
    asm volatile("s_wait_asynccnt 0x0" ::: "memory");
}

// ---------------------------------------------------------------------------
// fp32 -> f16 conversion (activations)
// ---------------------------------------------------------------------------
__global__ void cvt_f32_to_f16(const float* __restrict__ in, _Float16* __restrict__ out, int n) {
    int i = blockIdx.x * blockDim.x + threadIdx.x;
    if (i < n) out[i] = (_Float16)in[i];
}

// Key-bias vector over concatenated sequence: txt keys masked, img keys open.
__global__ void build_key_bias(const int* __restrict__ mask, float* __restrict__ bias) {
    int i = blockIdx.x * blockDim.x + threadIdx.x;
    if (i < S_ALL) bias[i] = (i < S_TXT) ? (mask[i] ? 0.0f : -1.0e30f) : 0.0f;
}

// ---------------------------------------------------------------------------
// Tiled WMMA GEMM:  C[M,N] = A[M,K](f16) * W[K,N](f32, converted) + bias
// Block tile 128x128, K-step 32. 8 waves in a 2x4 grid, each wave 64x32
// (4 M-frags x 2 N-frags => 8 v_wmma per K-step per wave).
// A tile staged via async-to-LDS DMA; W tile converted fp32->f16 in VGPRs
// (packed v_cvt + b32 stores) while the DMA runs.
// MODE 0: fp32 row-major output (outF).
// MODE 1: f16 head-major output [NH][S_ALL][HD] at row offset (outH).
// ---------------------------------------------------------------------------
template <int MODE>
__global__ __launch_bounds__(256) void gemm_wmma(
    const _Float16* __restrict__ A, const float* __restrict__ W,
    const float* __restrict__ bias, float* __restrict__ outF,
    _Float16* __restrict__ outH, int M, int N, int K, int rowOffset)
{
    __shared__ _Float16 As[128 * 32];   // [m][k] row-major
    __shared__ _Float16 Wt[128 * 32];   // [n][k] (transposed so B-frags are contiguous)

    const int tid   = threadIdx.x;
    const int wave  = tid >> 5, lane = tid & 31;
    const int laneM = lane & 15, sel = lane >> 4;
    const int waveM = wave >> 2, waveN = wave & 3;
    const int rowBase = blockIdx.y * 128, colBase = blockIdx.x * 128;

    v8f acc[4][2];
    #pragma unroll
    for (int mf = 0; mf < 4; ++mf)
        #pragma unroll
        for (int nf = 0; nf < 2; ++nf)
            #pragma unroll
            for (int r = 0; r < 8; ++r) acc[mf][nf][r] = 0.0f;

    const int arow = tid >> 1, aseg = tid & 1;   // A staging: 128 rows x 2 16-half segments
    const int wkp = tid & 15, wng = tid >> 4;    // W staging: k-pair 0..15, col-group 0..15

    const unsigned asLds = (unsigned)(size_t)As + (unsigned)(arow * 64 + aseg * 32);

    for (int k0 = 0; k0 < K; k0 += 32) {
        __syncthreads();
        {   // A tile: 128x32 f16, 32B per thread via 2x async b128 DMA
            unsigned long long g =
                (unsigned long long)(size_t)(A + (size_t)(rowBase + arow) * K + k0 + aseg * 16);
            async_b128(asLds, g);
            async_b128(asLds + 16, g + 16);
        }
        {   // W tile: 32x128 fp32 -> f16 transposed [n][k]; k-pairs -> packed b32 stores
            const float* s0 = W + (size_t)(k0 + 2 * wkp) * N + colBase + wng * 8;
            const float* s1 = s0 + N;
            #pragma unroll
            for (int i = 0; i < 8; ++i) {
                h2 v; v.x = (_Float16)s0[i]; v.y = (_Float16)s1[i];
                *(h2*)(Wt + (wng * 8 + i) * 32 + 2 * wkp) = v;
            }
        }
        if (k0 + 32 < K)  // gfx1250 global_prefetch for the next W tile
            __builtin_prefetch(W + (size_t)(k0 + 32 + 2 * wkp) * N + colBase + wng * 8, 0, 1);
        wait_async0();
        __syncthreads();

        v16h a[4], b[2];
        #pragma unroll
        for (int mf = 0; mf < 4; ++mf) {
            // 16-bit A 16x32 layout: lane<16 K{0..7,16..23}; lane>=16 K{8..15,24..31}
            const _Float16* rp = As + (waveM * 64 + mf * 16 + laneM) * 32 + sel * 8;
            a[mf] = ld_frag(rp, rp + 16);
        }
        #pragma unroll
        for (int nf = 0; nf < 2; ++nf) {
            // 16-bit B 32x16 layout: lanes 0-15 K{0..7,16..23}; lanes 16-31 K{8..15,24..31}
            const _Float16* rp = Wt + (waveN * 32 + nf * 16 + laneM) * 32 + sel * 8;
            b[nf] = ld_frag(rp, rp + 16);
        }
        #pragma unroll
        for (int mf = 0; mf < 4; ++mf)
            #pragma unroll
            for (int nf = 0; nf < 2; ++nf)
                acc[mf][nf] = __builtin_amdgcn_wmma_f32_16x16x32_f16(
                    false, a[mf], false, b[nf], (short)0, acc[mf][nf], false, false);
    }

    // C/D layout: VGPR r -> row r (lanes 0-15) / row r+8 (lanes 16-31), N = laneM.
    #pragma unroll
    for (int mf = 0; mf < 4; ++mf)
        #pragma unroll
        for (int nf = 0; nf < 2; ++nf)
            #pragma unroll
            for (int r = 0; r < 8; ++r) {
                int row = rowBase + waveM * 64 + mf * 16 + sel * 8 + r;
                int col = colBase + waveN * 32 + nf * 16 + laneM;
                float v = acc[mf][nf][r] + bias[col];
                if (MODE == 0) {
                    outF[(size_t)row * N + col] = v;
                } else {
                    int hh = col >> 7, dd = col & 127;
                    outH[((size_t)hh * S_ALL + rowOffset + row) * HD + dd] = (_Float16)v;
                }
            }
}

// ---------------------------------------------------------------------------
// Per-head RMSNorm + RoPE, fp32 in -> f16 head-major [NH][S_ALL][HD] out.
// One 128-thread block per (head, seq) row.
// ---------------------------------------------------------------------------
__global__ __launch_bounds__(128) void rmsnorm_rope(
    const float* __restrict__ in, const float* __restrict__ cosT,
    const float* __restrict__ sinT, const float* __restrict__ g,
    _Float16* __restrict__ outH, int rowOffset)
{
    __shared__ float red[128];
    __shared__ float xs[128];
    const int s = blockIdx.y, h = blockIdx.x, d = threadIdx.x;
    float x = in[(size_t)s * DIM + h * HD + d];
    red[d] = x * x;
    __syncthreads();
    #pragma unroll
    for (int off = 64; off > 0; off >>= 1) {
        if (d < off) red[d] += red[d + off];
        __syncthreads();
    }
    float rms = rsqrtf(red[0] * (1.0f / HD) + 1e-6f);
    xs[d] = x * rms * g[d];
    __syncthreads();
    int p = d >> 1;
    float c = cosT[s * (HD / 2) + p], sn = sinT[s * (HD / 2) + p];
    float x0 = xs[d & ~1], x1 = xs[d | 1];
    float o = ((d & 1) == 0) ? (x0 * c - x1 * sn) : (x0 * sn + x1 * c);
    outH[((size_t)h * S_ALL + rowOffset + s) * HD + d] = (_Float16)o;
}

// ---------------------------------------------------------------------------
// Flash attention, WMMA everywhere. Block = (head, 128-query tile), 8 waves;
// each wave owns 16 query rows. Q frags in registers, K tile DMA'd to LDS
// row-major (score B-frags contiguous), V tile transposed into LDS with packed
// b32 stores (PV B-frags contiguous), P re-laid out via a per-wave LDS pane.
// ---------------------------------------------------------------------------
__global__ __launch_bounds__(256) void attn_wmma(
    const _Float16* __restrict__ Q, const _Float16* __restrict__ Kh,
    const _Float16* __restrict__ V, const float* __restrict__ kbias,
    _Float16* __restrict__ out)
{
    __shared__ _Float16 Ks[128 * 128];      // [key][d]
    __shared__ _Float16 Vt[128 * 128];      // [d][key]
    __shared__ _Float16 Ps[8][16 * 128];    // per-wave P pane [m][key]

    const int h = blockIdx.y, qt = blockIdx.x;
    const int tid = threadIdx.x, wave = tid >> 5, lane = tid & 31;
    const int laneM = lane & 15, sel = lane >> 4;

    // Q fragments: 16 rows x 128 dim -> 4 K-chunks of 32
    v16h aq[4];
    {
        const _Float16* qb = Q + ((size_t)h * S_ALL + qt * 128 + wave * 16 + laneM) * HD;
        #pragma unroll
        for (int kc = 0; kc < 4; ++kc) {
            const _Float16* rp = qb + kc * 32 + sel * 8;
            aq[kc] = ld_frag(rp, rp + 16);
        }
    }

    v8f o[8];
    float m[8], l[8];
    #pragma unroll
    for (int r = 0; r < 8; ++r) { m[r] = -1.0e30f; l[r] = 0.0f; }
    #pragma unroll
    for (int nf = 0; nf < 8; ++nf)
        #pragma unroll
        for (int r = 0; r < 8; ++r) o[nf][r] = 0.0f;

    const float scale = 0.08838834764831845f;   // 1/sqrt(128)
    const unsigned ksLds = (unsigned)(size_t)Ks + (unsigned)tid * 16;

    for (int kt = 0; kt < S_ALL / 128; ++kt) {
        __syncthreads();
        {   // K tile: 32KB via async-to-LDS DMA (8 x b128 per thread)
            unsigned long long g =
                (unsigned long long)(size_t)(Kh + ((size_t)h * S_ALL + kt * 128) * HD)
                + (unsigned long long)tid * 16;
            #pragma unroll
            for (int i = 0; i < 8; ++i)
                async_b128(ksLds + 4096u * i, g + 4096ull * i);
        }
        {   // V tile transposed: key-pairs -> packed b32 stores (overlaps the DMA)
            const _Float16* vb = V + ((size_t)h * S_ALL + kt * 128) * HD;
            int kp = tid & 63, seg = tid >> 6;          // key pair, d-quarter
            const _Float16* v0 = vb + (2 * kp) * 128 + seg * 32;
            const _Float16* v1 = v0 + 128;
            #pragma unroll
            for (int i = 0; i < 32; ++i) {
                h2 p; p.x = v0[i]; p.y = v1[i];
                *(h2*)(Vt + (seg * 32 + i) * 128 + 2 * kp) = p;
            }
        }
        wait_async0();
        __syncthreads();

        // Scores: 16 q-rows x 128 keys = 8 N-frags, K=128 = 4 chained WMMAs each
        v8f sc[8];
        #pragma unroll
        for (int nf = 0; nf < 8; ++nf) {
            v8f c;
            #pragma unroll
            for (int r = 0; r < 8; ++r) c[r] = 0.0f;
            #pragma unroll
            for (int kc = 0; kc < 4; ++kc) {
                const _Float16* rp = Ks + (nf * 16 + laneM) * 128 + kc * 32 + sel * 8;
                v16h b = ld_frag(rp, rp + 16);
                c = __builtin_amdgcn_wmma_f32_16x16x32_f16(false, aq[kc], false, b,
                                                           (short)0, c, false, false);
            }
            float bb = kbias[kt * 128 + nf * 16 + laneM];
            #pragma unroll
            for (int r = 0; r < 8; ++r) c[r] = c[r] * scale + bb;
            sc[nf] = c;
        }

        // Online softmax. Row r / r+8 stats live per vgpr-slot; reduce across the
        // 16-lane half with shfl_xor (masks 1,2,4,8 stay inside the half).
        float mn[8];
        #pragma unroll
        for (int r = 0; r < 8; ++r) mn[r] = m[r];
        #pragma unroll
        for (int nf = 0; nf < 8; ++nf)
            #pragma unroll
            for (int r = 0; r < 8; ++r) mn[r] = fmaxf(mn[r], sc[nf][r]);
        #pragma unroll
        for (int off = 1; off <= 8; off <<= 1)
            #pragma unroll
            for (int r = 0; r < 8; ++r) mn[r] = fmaxf(mn[r], __shfl_xor(mn[r], off, 32));

        float rs[8];
        #pragma unroll
        for (int r = 0; r < 8; ++r) {
            float f = __expf(m[r] - mn[r]);
            l[r] *= f; m[r] = mn[r]; rs[r] = 0.0f;
            #pragma unroll
            for (int nf = 0; nf < 8; ++nf) o[nf][r] *= f;
        }
        #pragma unroll
        for (int nf = 0; nf < 8; ++nf)
            #pragma unroll
            for (int r = 0; r < 8; ++r) {
                float p = __expf(sc[nf][r] - m[r]);
                sc[nf][r] = p; rs[r] += p;
            }
        #pragma unroll
        for (int off = 1; off <= 8; off <<= 1)
            #pragma unroll
            for (int r = 0; r < 8; ++r) rs[r] += __shfl_xor(rs[r], off, 32);
        #pragma unroll
        for (int r = 0; r < 8; ++r) l[r] += rs[r];

        // P: C-layout -> f16 -> per-wave LDS pane -> A-layout frags
        _Float16* pw = Ps[wave];
        #pragma unroll
        for (int nf = 0; nf < 8; ++nf)
            #pragma unroll
            for (int r = 0; r < 8; ++r)
                pw[(sel * 8 + r) * 128 + nf * 16 + laneM] = (_Float16)sc[nf][r];

        // O += P(16 x 128keys) * V(128keys x 128d)
        #pragma unroll
        for (int nf = 0; nf < 8; ++nf)
            #pragma unroll
            for (int kc = 0; kc < 4; ++kc) {
                const _Float16* rp = pw + laneM * 128 + kc * 32 + sel * 8;
                v16h a = ld_frag(rp, rp + 16);
                const _Float16* vp = Vt + (nf * 16 + laneM) * 128 + kc * 32 + sel * 8;
                v16h b = ld_frag(vp, vp + 16);
                o[nf] = __builtin_amdgcn_wmma_f32_16x16x32_f16(false, a, false, b,
                                                               (short)0, o[nf], false, false);
            }
    }

    // Normalize and write row-major f16 [S_ALL][DIM] for the output projections.
    float inv[8];
    #pragma unroll
    for (int r = 0; r < 8; ++r) inv[r] = 1.0f / l[r];
    #pragma unroll
    for (int nf = 0; nf < 8; ++nf)
        #pragma unroll
        for (int r = 0; r < 8; ++r) {
            int row = qt * 128 + wave * 16 + sel * 8 + r;
            int col = h * HD + nf * 16 + laneM;
            out[(size_t)row * DIM + col] = (_Float16)(o[nf][r] * inv[r]);
        }
}

// ---------------------------------------------------------------------------
// Host-side orchestration
// ---------------------------------------------------------------------------
extern "C" void kernel_launch(void* const* d_in, const int* in_sizes, int n_in,
                              void* d_out, int out_size, void* d_ws, size_t ws_size,
                              hipStream_t stream)
{
    const float* x_img   = (const float*)d_in[0];
    const float* x_txt   = (const float*)d_in[1];
    const int*   emask   = (const int*)  d_in[2];
    const float* img_cos = (const float*)d_in[3];
    const float* img_sin = (const float*)d_in[4];
    const float* txt_cos = (const float*)d_in[5];
    const float* txt_sin = (const float*)d_in[6];
    const float* wq  = (const float*)d_in[7];   const float* bq  = (const float*)d_in[8];
    const float* wk  = (const float*)d_in[9];   const float* bk  = (const float*)d_in[10];
    const float* wv  = (const float*)d_in[11];  const float* bv  = (const float*)d_in[12];
    const float* waq = (const float*)d_in[13];  const float* baq = (const float*)d_in[14];
    const float* wak = (const float*)d_in[15];  const float* bak = (const float*)d_in[16];
    const float* wav = (const float*)d_in[17];  const float* bav = (const float*)d_in[18];
    const float* wo  = (const float*)d_in[19];  const float* bo  = (const float*)d_in[20];
    const float* wao = (const float*)d_in[21];  const float* bao = (const float*)d_in[22];
    const float* gq  = (const float*)d_in[23];  const float* gk  = (const float*)d_in[24];
    const float* gaq = (const float*)d_in[25];  const float* gak = (const float*)d_in[26];

    // Workspace carve-up (~128 MB, 256B-aligned chunks)
    char* base = (char*)d_ws;
    size_t off = 0;
    auto carve = [&](size_t bytes) -> char* {
        char* p = base + off;
        off += (bytes + 255) & ~(size_t)255;
        return p;
    };
    _Float16* Xi  = (_Float16*)carve((size_t)S_IMG * DIM * 2);
    _Float16* Xt  = (_Float16*)carve((size_t)S_TXT * DIM * 2);
    float*    qf  = (float*)   carve((size_t)S_ALL * DIM * 4);
    float*    kf  = (float*)   carve((size_t)S_ALL * DIM * 4);
    _Float16* qh  = (_Float16*)carve((size_t)NH * S_ALL * HD * 2);
    _Float16* kh  = (_Float16*)carve((size_t)NH * S_ALL * HD * 2);
    _Float16* vh  = (_Float16*)carve((size_t)NH * S_ALL * HD * 2);
    _Float16* ao  = (_Float16*)carve((size_t)S_ALL * DIM * 2);
    float*    kbs = (float*)   carve((size_t)S_ALL * 4);

    // 1) activations -> f16, key bias
    cvt_f32_to_f16<<<(S_IMG * DIM + 255) / 256, 256, 0, stream>>>(x_img, Xi, S_IMG * DIM);
    cvt_f32_to_f16<<<(S_TXT * DIM + 255) / 256, 256, 0, stream>>>(x_txt, Xt, S_TXT * DIM);
    build_key_bias<<<(S_ALL + 255) / 256, 256, 0, stream>>>(emask, kbs);

    dim3 gI(DIM / 128, S_IMG / 128), gT(DIM / 128, S_TXT / 128);

    // 2) Q/K projections (fp32 out, pre-norm), V projection (f16 head-major)
    gemm_wmma<0><<<gI, 256, 0, stream>>>(Xi, wq, bq, qf + (size_t)S_TXT * DIM, nullptr, S_IMG, DIM, DIM, 0);
    gemm_wmma<0><<<gT, 256, 0, stream>>>(Xt, waq, baq, qf, nullptr, S_TXT, DIM, DIM, 0);
    gemm_wmma<0><<<gI, 256, 0, stream>>>(Xi, wk, bk, kf + (size_t)S_TXT * DIM, nullptr, S_IMG, DIM, DIM, 0);
    gemm_wmma<0><<<gT, 256, 0, stream>>>(Xt, wak, bak, kf, nullptr, S_TXT, DIM, DIM, 0);
    gemm_wmma<1><<<gI, 256, 0, stream>>>(Xi, wv, bv, nullptr, vh, S_IMG, DIM, DIM, S_TXT);
    gemm_wmma<1><<<gT, 256, 0, stream>>>(Xt, wav, bav, nullptr, vh, S_TXT, DIM, DIM, 0);

    // 3) RMSNorm + RoPE into head-major f16
    rmsnorm_rope<<<dim3(NH, S_IMG), 128, 0, stream>>>(qf + (size_t)S_TXT * DIM, img_cos, img_sin, gq,  qh, S_TXT);
    rmsnorm_rope<<<dim3(NH, S_TXT), 128, 0, stream>>>(qf,                        txt_cos, txt_sin, gaq, qh, 0);
    rmsnorm_rope<<<dim3(NH, S_IMG), 128, 0, stream>>>(kf + (size_t)S_TXT * DIM, img_cos, img_sin, gk,  kh, S_TXT);
    rmsnorm_rope<<<dim3(NH, S_TXT), 128, 0, stream>>>(kf,                        txt_cos, txt_sin, gak, kh, 0);

    // 4) Flash attention
    attn_wmma<<<dim3(S_ALL / 128, NH), 256, 0, stream>>>(qh, kh, vh, kbs, ao);

    // 5) Output projections straight into d_out (img first, then txt)
    gemm_wmma<0><<<gI, 256, 0, stream>>>(ao + (size_t)S_TXT * DIM, wo, bo, (float*)d_out, nullptr, S_IMG, DIM, DIM, 0);
    gemm_wmma<0><<<gT, 256, 0, stream>>>(ao, wao, bao, (float*)d_out + (size_t)S_IMG * DIM, nullptr, S_TXT, DIM, DIM, 0);
}
